// RGCNCDBaseline_27685359190065
// MI455X (gfx1250) — matrix operand order repaired and checked by reading, not compiled, not run on hardware
//
#include <hip/hip_runtime.h>
#include <hip/hip_bf16.h>

typedef __attribute__((ext_vector_type(16))) _Float16 v16h;
typedef __attribute__((ext_vector_type(8)))  _Float16 v8h;
typedef __attribute__((ext_vector_type(4)))  _Float16 v4h;
typedef __attribute__((ext_vector_type(8)))  float    v8f;

#define H_DIM 128
#define HH    (H_DIM * H_DIM)
#define N_CHEM 60000
#define N_DIS  40000
#define N_TOT  (N_CHEM + N_DIS)

// ---------------------------------------------------------------------------
// WMMA layouts (wave32, v_wmma_f32_16x16x32_f16):
//  A 16x32 : lane L holds row M=L&15; halves 0..7 -> K=8*(L>=16)+{0..7},
//            halves 8..15 -> K=8*(L>=16)+{16..23}.
//  B 32x16 : lane L holds col N=L&15; half h -> K=8*(L>=16)+(h&7)+16*(h>>3).
//  C/D f32 : vgpr v, lane L -> (M = v + 8*(L>=16), N = L&15).
// Weights are pre-packed so a B tile is one contiguous v16h per lane.
// ---------------------------------------------------------------------------

// A tile from f16 row-major features: two contiguous 16B loads.
__device__ __forceinline__ v16h load_a16(const _Float16* __restrict__ p, int hi) {
  const _Float16* q = p + 8 * hi;
  v8h lo = *(const v8h*)(q);
  v8h hh = *(const v8h*)(q + 16);
  v16h a;
#pragma unroll
  for (int i = 0; i < 8; ++i) { a[i] = lo[i]; a[8 + i] = hh[i]; }
  return a;
}

// A tile from the f32 segment-sum buffer, scaled by 1/deg while converting.
__device__ __forceinline__ v16h load_a32_scaled(const float* __restrict__ p, int hi,
                                                float s) {
  const float* q = p + 8 * hi;
  v16h a;
#pragma unroll
  for (int i = 0; i < 8; ++i) a[i]     = (_Float16)(q[i] * s);
#pragma unroll
  for (int i = 0; i < 8; ++i) a[8 + i] = (_Float16)(q[16 + i] * s);
  return a;
}

// ---------------------------------------------------------------------------
// One-time packing of an HxH f32 weight into B-operand-ready f16 layout:
// half-index i = ((tile*32 + lane)*16 + h), tile = kc*8 + ct.
// ---------------------------------------------------------------------------
__global__ void rgcn_pack_b(const float* __restrict__ W, _Float16* __restrict__ out) {
  int i = blockIdx.x * blockDim.x + threadIdx.x;
  if (i >= HH) return;
  int h    = i & 15;
  int lane = (i >> 4) & 31;
  int tile = i >> 9;
  int kc = tile >> 3, ct = tile & 7;
  int k = kc * 32 + 8 * (lane >> 4) + (h & 7) + 16 * (h >> 3);
  int n = ct * 16 + (lane & 15);
  out[i] = (_Float16)W[(size_t)k * H_DIM + n];
}

__global__ void rgcn_zero_f32(float* __restrict__ p, size_t n) {
  size_t i = (size_t)blockIdx.x * blockDim.x + threadIdx.x;
  size_t stride = (size_t)gridDim.x * blockDim.x;
  for (; i < n; i += stride) p[i] = 0.0f;
}

__global__ void rgcn_cvt_f16(const float* __restrict__ in, _Float16* __restrict__ out,
                             size_t n) {
  size_t i = (size_t)blockIdx.x * blockDim.x + threadIdx.x;
  size_t stride = (size_t)gridDim.x * blockDim.x;
  for (; i < n; i += stride) out[i] = (_Float16)in[i];
}

// In-degree per node (dst halves hit disjoint node ranges -> one buffer).
__global__ void rgcn_degree(const int* __restrict__ dst, int ne2,
                            float* __restrict__ cnt) {
  int i = blockIdx.x * blockDim.x + threadIdx.x;
  if (i < ne2) atomicAdd(&cnt[dst[i]], 1.0f);
}

// Segment-sum of f16 feature rows into f32 accumulator. One wave per edge:
// 32 lanes x 4 halves (8B load) covers the 128-wide row; 4 f32 atomics/lane.
__global__ __launch_bounds__(256) void rgcn_scatter(const _Float16* __restrict__ x16,
                                                    const int* __restrict__ src,
                                                    const int* __restrict__ dst,
                                                    float* __restrict__ S, int ne2) {
  int gid  = blockIdx.x * blockDim.x + threadIdx.x;
  int e    = gid >> 5;
  int lane = gid & 31;
  if (e >= ne2) return;
  int s = src[e];
  int d = dst[e];
  v4h v = ((const v4h*)(x16 + (size_t)s * H_DIM))[lane];
  float* Sp = S + (size_t)d * H_DIM + lane * 4;
  atomicAdd(Sp + 0, (float)v[0]);
  atomicAdd(Sp + 1, (float)v[1]);
  atomicAdd(Sp + 2, (float)v[2]);
  atomicAdd(Sp + 3, (float)v[3]);
}

// ---------------------------------------------------------------------------
// Fused layer: xout = f16( relu?( x @ Wroot + (S/max(deg,1)) @ Wr[region] + b ) )
// One wave computes a 16x32 output strip (two accumulators): A loads are
// reused across both column tiles, 4 WMMAs per K-chunk on 2 indep chains.
// ---------------------------------------------------------------------------
__global__ __launch_bounds__(256) void rgcn_gemm(const _Float16* __restrict__ x16,
                                                 const float* __restrict__ S,
                                                 const float* __restrict__ cnt,
                                                 const v16h* __restrict__ Broot,  // packed HxH
                                                 const v16h* __restrict__ Brel0,  // packed HxH
                                                 const v16h* __restrict__ Brel1,  // packed HxH
                                                 const float* __restrict__ bias,  // H
                                                 _Float16* __restrict__ xout,
                                                 int do_relu) {
  const int wave = (int)((blockIdx.x * blockDim.x + threadIdx.x) >> 5);
  const int lane = threadIdx.x & 31;
  const int numWaves = (N_TOT / 16) * 4;   // 2 column tiles per wave
  if (wave >= numWaves) return;

  const int rt    = wave >> 2;
  const int cpair = wave & 3;
  const int row0  = rt * 16;
  const int ct0   = cpair * 2;
  const int r     = lane & 15;
  const int hi    = lane >> 4;
  const int row   = row0 + r;

  // N_CHEM % 16 == 0 -> tiles are region-homogeneous.
  const v16h* Brel = (row0 < N_CHEM) ? Brel1 : Brel0;

  const float inv = __builtin_amdgcn_rcpf(fmaxf(cnt[row], 1.0f));
  const _Float16* xrow = x16 + (size_t)row * H_DIM;
  const float*    srow = S   + (size_t)row * H_DIM;

  v8f acc0 = {}, acc1 = {};
#pragma unroll
  for (int kc = 0; kc < 4; ++kc) {
    const int k0 = kc * 32;
    const int t0 = kc * 8 + ct0;
    v16h ax = load_a16(xrow + k0, hi);
    v16h as = load_a32_scaled(srow + k0, hi, inv);
    v16h b;
    b = Broot[(size_t)t0 * 32 + lane];
    acc0 = __builtin_amdgcn_wmma_f32_16x16x32_f16(false, ax, false, b, (short)0, acc0, false, false);
    b = Brel[(size_t)t0 * 32 + lane];
    acc0 = __builtin_amdgcn_wmma_f32_16x16x32_f16(false, as, false, b, (short)0, acc0, false, false);
    b = Broot[(size_t)(t0 + 1) * 32 + lane];
    acc1 = __builtin_amdgcn_wmma_f32_16x16x32_f16(false, ax, false, b, (short)0, acc1, false, false);
    b = Brel[(size_t)(t0 + 1) * 32 + lane];
    acc1 = __builtin_amdgcn_wmma_f32_16x16x32_f16(false, as, false, b, (short)0, acc1, false, false);
  }

  const float bv0 = bias[ct0 * 16 + r];
  const float bv1 = bias[ct0 * 16 + 16 + r];
#pragma unroll
  for (int v = 0; v < 8; ++v) {
    const size_t orow = (size_t)(row0 + v + 8 * hi) * H_DIM;
    float v0 = acc0[v] + bv0;
    float v1 = acc1[v] + bv1;
    if (do_relu) { v0 = fmaxf(v0, 0.0f); v1 = fmaxf(v1, 0.0f); }
    xout[orow + ct0 * 16 + r]      = (_Float16)v0;
    xout[orow + ct0 * 16 + 16 + r] = (_Float16)v1;
  }
}

// ---------------------------------------------------------------------------
// Scoring: out[b] = sum_j (c_b @ Wscore)[j] * d_b[j].
// One wave per 16-query tile; A rows held in registers across all 8 column
// blocks; elementwise multiply by gathered d in C-layout; 16-lane reduction.
// ---------------------------------------------------------------------------
__global__ __launch_bounds__(128) void rgcn_score(const _Float16* __restrict__ x16,
                                                  const v16h* __restrict__ Bscore,
                                                  const int* __restrict__ chem_ids,
                                                  const int* __restrict__ dis_ids,
                                                  float* __restrict__ out, int B) {
  const int wave = (int)((blockIdx.x * blockDim.x + threadIdx.x) >> 5);
  const int lane = threadIdx.x & 31;
  const int numTiles = B / 16;
  if (wave >= numTiles) return;

  const int q0 = wave * 16;
  const int r  = lane & 15;
  const int hi = lane >> 4;

  const _Float16* crow = x16 + (size_t)chem_ids[q0 + r] * H_DIM;
  v16h aT[4];
#pragma unroll
  for (int kc = 0; kc < 4; ++kc) aT[kc] = load_a16(crow + kc * 32, hi);

  int drow[8];
#pragma unroll
  for (int v = 0; v < 8; ++v)
    drow[v] = N_CHEM + dis_ids[q0 + v + 8 * hi];

  float p[8] = {0.f, 0.f, 0.f, 0.f, 0.f, 0.f, 0.f, 0.f};
#pragma unroll
  for (int jb = 0; jb < 8; ++jb) {
    v8f acc = {};
#pragma unroll
    for (int kc = 0; kc < 4; ++kc) {
      v16h b = Bscore[(size_t)(kc * 8 + jb) * 32 + lane];
      acc = __builtin_amdgcn_wmma_f32_16x16x32_f16(false, aT[kc], false, b,
                                                   (short)0, acc, false, false);
    }
#pragma unroll
    for (int v = 0; v < 8; ++v) {
      float dv = (float)x16[(size_t)drow[v] * H_DIM + jb * 16 + r];
      p[v] += acc[v] * dv;
    }
  }

  // Row m = v + 8*hi is spread over the 16 lanes of this half in vgpr v.
#pragma unroll
  for (int m = 1; m < 16; m <<= 1) {
#pragma unroll
    for (int v = 0; v < 8; ++v) p[v] += __shfl_xor(p[v], m, 32);
  }

  if (r == 0) {
#pragma unroll
    for (int v = 0; v < 8; ++v) out[q0 + v + 8 * hi] = p[v];
  }
}

// ---------------------------------------------------------------------------
// Host launcher
// ---------------------------------------------------------------------------
extern "C" void kernel_launch(void* const* d_in, const int* in_sizes, int n_in,
                              void* d_out, int out_size, void* d_ws, size_t ws_size,
                              hipStream_t stream) {
  const float* node_emb = (const float*)d_in[0];   // N x H
  const float* Wrel     = (const float*)d_in[1];   // L x 2 x H x H
  const float* Wroot    = (const float*)d_in[2];   // L x H x H
  const float* bias     = (const float*)d_in[3];   // L x H
  const float* Wscore   = (const float*)d_in[4];   // H x H
  const int*   src      = (const int*)d_in[5];     // 2E
  const int*   dst      = (const int*)d_in[6];     // 2E
  const int*   chem_ids = (const int*)d_in[7];     // B
  const int*   dis_ids  = (const int*)d_in[8];     // B
  const int ne2 = in_sizes[5];
  const int B   = in_sizes[7];
  float* out = (float*)d_out;

  // Workspace: S (N*H f32) | cnt (N f32) | x16A (N*H f16) | x16B | Wpk (7*H*H f16)
  const size_t nh = (size_t)N_TOT * H_DIM;
  float*    S    = (float*)d_ws;
  float*    cnt  = S + nh;
  _Float16* x16A = (_Float16*)(cnt + N_TOT);
  _Float16* x16B = x16A + nh;
  _Float16* Wpk  = x16B + nh;
  // Packed matrix order: [Wroot0, Wrel0_r0, Wrel0_r1, Wroot1, Wrel1_r0, Wrel1_r1, Wscore]
  _Float16* pk[7];
  for (int m = 0; m < 7; ++m) pk[m] = Wpk + (size_t)m * HH;

  const int packBlocks = (HH + 255) / 256;
  rgcn_pack_b<<<packBlocks, 256, 0, stream>>>(Wroot,            pk[0]);
  rgcn_pack_b<<<packBlocks, 256, 0, stream>>>(Wrel,             pk[1]);
  rgcn_pack_b<<<packBlocks, 256, 0, stream>>>(Wrel + HH,        pk[2]);
  rgcn_pack_b<<<packBlocks, 256, 0, stream>>>(Wroot + HH,       pk[3]);
  rgcn_pack_b<<<packBlocks, 256, 0, stream>>>(Wrel + 2 * HH,    pk[4]);
  rgcn_pack_b<<<packBlocks, 256, 0, stream>>>(Wrel + 3 * HH,    pk[5]);
  rgcn_pack_b<<<packBlocks, 256, 0, stream>>>(Wscore,           pk[6]);

  rgcn_zero_f32<<<2048, 256, 0, stream>>>(S, nh + N_TOT);      // S + cnt contiguous
  rgcn_degree<<<(ne2 + 255) / 256, 256, 0, stream>>>(dst, ne2, cnt);
  rgcn_cvt_f16<<<2048, 256, 0, stream>>>(node_emb, x16A, nh);

  const int gemmWaves  = (N_TOT / 16) * 4;                     // 25000
  const int gemmBlocks = (gemmWaves + 7) / 8;

  const _Float16* xcur = x16A;
  _Float16* xnext = x16B;
  for (int l = 0; l < 2; ++l) {
    if (l > 0) rgcn_zero_f32<<<2048, 256, 0, stream>>>(S, nh);
    {
      long long threads = (long long)ne2 * 32;
      int blocks = (int)((threads + 255) / 256);
      rgcn_scatter<<<blocks, 256, 0, stream>>>(xcur, src, dst, S, ne2);
    }
    rgcn_gemm<<<gemmBlocks, 256, 0, stream>>>(
        xcur, S, cnt,
        (const v16h*)pk[3 * l + 0], (const v16h*)pk[3 * l + 1], (const v16h*)pk[3 * l + 2],
        bias + (size_t)l * H_DIM,
        xnext, (l + 1 < 2) ? 1 : 0);
    xcur  = xnext;
    xnext = x16A;
  }

  const int scoreBlocks = (B / 16 + 3) / 4;                    // 4 waves / 128-thr block
  rgcn_score<<<scoreBlocks, 128, 0, stream>>>(xcur, (const v16h*)pk[6],
                                              chem_ids, dis_ids, out, B);
}
